// MCB_CoAtt_45148696216204
// MI455X (gfx1250) — compile-verified, matrix operand order checked by reading, not run on hardware
//
#include <hip/hip_runtime.h>
#include <stddef.h>

// ---------------------------------------------------------------------------
// MCB co-attention VQA pipeline for MI455X (gfx1250, wave32, WMMA + TDM).
// All GEMMs: v_wmma_f32_16x16x32_bf16, operands pre-converted to zero-padded
// bf16 so the inner loop is pure global_load_b128 + wmma (no guards, no cvt).
// Each wave computes a 16x64 output strip (1 A frag -> 4 WMMAs).
// MCB1 conv: sparse-q gather over dense folded image sketch (62.7MB, fits the
// 192MB L2). MCB2 conv: from LDS, with the A-sketch row staged into LDS via
// the Tensor Data Mover (tensor_load_to_lds + s_wait_tensorcnt) when the
// toolchain exposes it. Workspace: ~225 MB of d_ws.
// ---------------------------------------------------------------------------

typedef __attribute__((ext_vector_type(16))) __bf16 v16bf;
typedef __attribute__((ext_vector_type(8)))  float  v8f;
typedef __attribute__((ext_vector_type(4)))  unsigned int u32x4;
typedef __attribute__((ext_vector_type(8)))  int i32x8;
typedef __attribute__((ext_vector_type(4)))  int i32x4;

#if __has_builtin(__builtin_amdgcn_tensor_load_to_lds) && __has_builtin(__builtin_amdgcn_s_wait_tensorcnt)
#define USE_TDM 1
#else
#define USE_TDM 0
#endif

#define Bn    64
#define Tn    26
#define En    300
#define EnP   320     // En padded to mult of 32
#define Hn    1024
#define G4n   4096
#define FEATn 2048
#define Sn    49
#define D1n   245000
#define D2n   5000
#define MOUTn 1000
#define MOUTP 1024    // MOUTn padded
#define VOCn  3000
#define VOCP  3008    // VOCn padded to mult of 16

__device__ __forceinline__ unsigned short f2bf(float f) {
  unsigned int u = __float_as_uint(f);
  unsigned int r = 0x7FFFu + ((u >> 16) & 1u);
  return (unsigned short)((u + r) >> 16);
}

__device__ __forceinline__ float sigf(float x) { return 1.f / (1.f + __expf(-x)); }

// ------------------------------- utility -----------------------------------
__global__ void k_zero(float* p, size_t n) {
  size_t i = (size_t)blockIdx.x * 256 + threadIdx.x;
  if (i < n) p[i] = 0.f;
}
__global__ void k_zero_us(unsigned short* p, size_t n) {
  size_t i = (size_t)blockIdx.x * 256 + threadIdx.x;
  if (i < n) p[i] = 0;
}

// f32 (rows_src x K) -> bf16 (rows_dst x Kp), zero padded in K and rows.
__global__ void k_cvt_bf(const float* __restrict__ src, unsigned short* __restrict__ dst,
                         int rows_src, int K, int Kp, long total) {
  long idx = (long)blockIdx.x * 256 + threadIdx.x;
  if (idx >= total) return;
  int r = (int)(idx / Kp), k = (int)(idx % Kp);
  float v = (r < rows_src && k < K) ? src[(size_t)r * K + k] : 0.f;
  dst[idx] = f2bf(v);
}

// ques_embed (B,T,E) -> bf16 rows rr = t*64+b, Kp=320
__global__ void k_cvt_qe(const float* __restrict__ qe, unsigned short* __restrict__ dst) {
  int idx = blockIdx.x * 256 + threadIdx.x;  // < 1664*320
  if (idx >= (Tn * Bn) * EnP) return;
  int r = idx / EnP, k = idx % EnP;
  int t = r >> 6, b = r & 63;
  float v = (k < En) ? qe[((size_t)(b * Tn + t)) * En + k] : 0.f;
  dst[idx] = f2bf(v);
}

// iq (B,1000,49) -> bf16 rows rr = b*49+s, cols m, Kp=1024
__global__ void k_cvt_iqT(const float* __restrict__ iq, unsigned short* __restrict__ dst) {
  int idx = blockIdx.x * 256 + threadIdx.x;  // < 3136*1024
  if (idx >= (Bn * Sn) * MOUTP) return;
  int r = idx / MOUTP, m = idx % MOUTP;
  int b = r / Sn, s = r % Sn;
  float v = (m < MOUTn) ? iq[(size_t)b * (MOUTn * Sn) + (size_t)m * Sn + s] : 0.f;
  dst[idx] = f2bf(v);
}

// ------------------------- WMMA GEMM (bf16 operands) -----------------------
// C[row, col] = sum_k A_bf[row*Kp+k] * B_bf[col*Kp+k]  (both K-major bf16)
// One wave computes a 16x64 strip: 1 A fragment feeds 4 WMMAs.
// A/C fragment layouts per ISA 7.12.2; B assumed transpose-symmetric to A.
__global__ void k_gemm_bf4(const unsigned short* __restrict__ A,
                           const unsigned short* __restrict__ Bm,
                           float* __restrict__ C,
                           int Kp, int ldc, int Nvalid,
                           const float* __restrict__ bias0,
                           const float* __restrict__ bias1,
                           int relu) {
  int row0 = blockIdx.x * 16;
  int col0 = blockIdx.y * 64;
  int lane = threadIdx.x & 31;
  int rc = lane & 15, kb = (lane >> 4) * 8;
  const unsigned short* ap  = A  + (size_t)(row0 + rc) * Kp + kb;
  const unsigned short* bp  = Bm + (size_t)(col0 + rc) * Kp + kb;
  size_t bstep = (size_t)16 * Kp;
  v8f acc0 = {}, acc1 = {}, acc2 = {}, acc3 = {};
  for (int k0 = 0; k0 < Kp; k0 += 32) {
    union { v16bf v; uint4 q[2]; } fa, f0, f1, f2, f3;
    fa.q[0] = *(const uint4*)(ap + k0);
    fa.q[1] = *(const uint4*)(ap + k0 + 16);
    f0.q[0] = *(const uint4*)(bp + k0);
    f0.q[1] = *(const uint4*)(bp + k0 + 16);
    f1.q[0] = *(const uint4*)(bp + bstep + k0);
    f1.q[1] = *(const uint4*)(bp + bstep + k0 + 16);
    f2.q[0] = *(const uint4*)(bp + 2 * bstep + k0);
    f2.q[1] = *(const uint4*)(bp + 2 * bstep + k0 + 16);
    f3.q[0] = *(const uint4*)(bp + 3 * bstep + k0);
    f3.q[1] = *(const uint4*)(bp + 3 * bstep + k0 + 16);
    acc0 = __builtin_amdgcn_wmma_f32_16x16x32_bf16(false, fa.v, false, f0.v, (short)0, acc0, false, false);
    acc1 = __builtin_amdgcn_wmma_f32_16x16x32_bf16(false, fa.v, false, f1.v, (short)0, acc1, false, false);
    acc2 = __builtin_amdgcn_wmma_f32_16x16x32_bf16(false, fa.v, false, f2.v, (short)0, acc2, false, false);
    acc3 = __builtin_amdgcn_wmma_f32_16x16x32_bf16(false, fa.v, false, f3.v, (short)0, acc3, false, false);
  }
  int mo = (lane >> 4) * 8;
#pragma unroll
  for (int i = 0; i < 4; ++i) {
    v8f a = (i == 0) ? acc0 : (i == 1) ? acc1 : (i == 2) ? acc2 : acc3;
    int g = col0 + 16 * i + rc;
    if (g < Nvalid) {
      float badd = (bias0 ? bias0[g] : 0.f) + (bias1 ? bias1[g] : 0.f);
#pragma unroll
      for (int v = 0; v < 8; ++v) {
        float val = a[v] + badd;
        if (relu) val = fmaxf(val, 0.f);
        C[(size_t)(row0 + mo + v) * ldc + g] = val;
      }
    }
  }
}

// ------------------------------ LSTM gates ---------------------------------
__global__ void k_gates(const float* __restrict__ xW, const float* __restrict__ gbuf,
                        float* __restrict__ cst, float* __restrict__ hs,
                        unsigned short* __restrict__ hsA, unsigned short* __restrict__ hsB,
                        int t) {
  int idx = blockIdx.x * blockDim.x + threadIdx.x;  // < 64*1024
  int b = idx >> 10, hh = idx & 1023;
  const float* xr = xW + ((size_t)(t * Bn + b)) * G4n;
  const float* gr = gbuf + (size_t)b * G4n;
  float gi = xr[hh]        + gr[hh];
  float gf = xr[1024 + hh] + gr[1024 + hh];
  float gg = xr[2048 + hh] + gr[2048 + hh];
  float go = xr[3072 + hh] + gr[3072 + hh];
  float cn = sigf(gf) * cst[idx] + sigf(gi) * tanhf(gg);
  cst[idx] = cn;
  float hv = sigf(go) * tanhf(cn);
  hs[(size_t)t * (Bn * Hn) + idx] = hv;
  unsigned short hb = f2bf(hv);
  hsA[(size_t)t * (Bn * Hn) + idx] = hb;                   // rows rr = t*64+b
  hsB[((size_t)(b * Tn + t)) * Hn + hh] = hb;              // rows rr = b*26+t
}

// ------------------------- attention small kernels -------------------------
__global__ void k_att_logits(const float* __restrict__ act, const float* __restrict__ W2,
                             const float* __restrict__ b2, float* __restrict__ outb, int nt) {
  int idx = blockIdx.x * blockDim.x + threadIdx.x;
  int total = Bn * 2 * nt;
  if (idx >= total) return;
  int t = idx % nt; int bg = idx / nt; int g = bg & 1; int b = bg >> 1;
  const float* a = act + ((size_t)(b * nt + t)) * 512;
  const float* w = W2 + g * 512;
  float s = b2[g];
  for (int o = 0; o < 512; ++o) s += w[o] * a[o];
  outb[idx] = s;
}

__global__ void k_softmax_small(float* __restrict__ buf, int rows, int n) {
  int r = blockIdx.x * blockDim.x + threadIdx.x;
  if (r >= rows) return;
  float* p = buf + (size_t)r * n;
  float mx = -1e30f;
  for (int i = 0; i < n; ++i) mx = fmaxf(mx, p[i]);
  float s = 0.f;
  for (int i = 0; i < n; ++i) { float e = __expf(p[i] - mx); p[i] = e; s += e; }
  float inv = 1.f / s;
  for (int i = 0; i < n; ++i) p[i] *= inv;
}

__global__ void k_qfeat(const float* __restrict__ qatt, const float* __restrict__ hs,
                        float* __restrict__ qf) {
  int idx = blockIdx.x * 256 + threadIdx.x;  // < 64*2048
  int b = idx >> 11; int r = idx & 2047; int g = r >> 10; int h = r & 1023;
  const float* at = qatt + (size_t)(b * 2 + g) * Tn;
  float s = 0.f;
  for (int t = 0; t < Tn; ++t) s += at[t] * hs[(size_t)t * (Bn * Hn) + b * Hn + h];
  qf[idx] = s;
}

__global__ void k_ifeat(const float* __restrict__ iatt, const float* __restrict__ img,
                        float* __restrict__ ift) {
  int idx = blockIdx.x * 256 + threadIdx.x;  // < 64*4096
  int b = idx >> 12; int r = idx & 4095; int g = r >> 11; int f = r & 2047;
  const float* at = iatt + (size_t)(b * 2 + g) * Sn;
  float s = 0.f;
  for (int sp = 0; sp < Sn; ++sp) s += at[sp] * img[((size_t)(b * Sn + sp)) * FEATn + f];
  ift[idx] = s;
}

// ------------------------------ MCB stage 1 --------------------------------
__global__ void k_sketch1(const float* __restrict__ img, const int* __restrict__ h1x,
                          const int* __restrict__ s1x, float* __restrict__ A1) {
  size_t idx = (size_t)blockIdx.x * 256 + threadIdx.x;  // over B*S*FEAT (img layout)
  if (idx >= (size_t)Bn * Sn * FEATn) return;
  int b = (int)(idx / (Sn * FEATn)); int rem = (int)(idx % (Sn * FEATn));
  int s = rem / FEATn, f = rem % FEATn;
  int n = f * Sn + s;  // flat index of img.reshape(B, FEAT*S)
  float v = img[idx] * (float)(2 * s1x[n] - 1);
  atomicAdd(&A1[(size_t)b * D1n + h1x[n]], v);
}

// Afold[b,k] = sum_{f<5} A1[b,(k+49f) mod D1]  (pre-folds FACTOR group-sum)
__global__ void k_fold1(const float* __restrict__ A1, float* __restrict__ Af) {
  size_t idx = (size_t)blockIdx.x * 256 + threadIdx.x;
  if (idx >= (size_t)Bn * D1n) return;
  int b = (int)(idx / D1n); int k = (int)(idx % D1n);
  const float* Ab = A1 + (size_t)b * D1n;
  float s = 0.f;
#pragma unroll
  for (int f = 0; f < 5; ++f) { int kk = k + 49 * f; if (kk >= D1n) kk -= D1n; s += Ab[kk]; }
  Af[idx] = s;
}

// iq[b,m,s] = sum_j qv[b,j] * Afold[b,(m*245+s-p_j) mod D1]
__global__ void k_conv1(const float* __restrict__ Af, const float* __restrict__ qf,
                        const int* __restrict__ h1q, const int* __restrict__ s1q,
                        float* __restrict__ iq) {
  __shared__ int   ps[2048];
  __shared__ float qv[2048];
  int m = blockIdx.x, b = blockIdx.y;
  for (int j = threadIdx.x; j < 2048; j += 64) {
    ps[j] = h1q[j];
    qv[j] = qf[(size_t)b * 2048 + j] * (float)(2 * s1q[j] - 1);
  }
  __syncthreads();
  int s = threadIdx.x;
  if (s < Sn) {
    const float* Ab = Af + (size_t)b * D1n;
    int k0 = m * 245 + s;
    float sum = 0.f;
    for (int j = 0; j < 2048; ++j) {
      int a = k0 - ps[j]; if (a < 0) a += D1n;
      if ((j & 63) == 0 && j + 64 < 2048) {
        int an = k0 - ps[j + 64]; if (an < 0) an += D1n;
        __builtin_prefetch(Ab + an, 0, 0);   // global_prefetch_b8
      }
      sum += qv[j] * Ab[a];
    }
    iq[(size_t)b * (MOUTn * Sn) + m * Sn + s] = sum;
  }
}

// signed-sqrt then row-wise L2 normalize (len elements per block/batch)
__global__ void k_ssqrt_l2(float* __restrict__ buf, int len) {
  int b = blockIdx.x;
  float* p = buf + (size_t)b * len;
  __shared__ float red[256];
  float acc = 0.f;
  for (int i = threadIdx.x; i < len; i += 256) {
    float v = p[i];
    float sv = (v >= 0.f) ? sqrtf(v) : -sqrtf(-v);
    p[i] = sv;
    acc += sv * sv;
  }
  red[threadIdx.x] = acc; __syncthreads();
  for (int o = 128; o > 0; o >>= 1) {
    if (threadIdx.x < o) red[threadIdx.x] += red[threadIdx.x + o];
    __syncthreads();
  }
  float inv = 1.f / fmaxf(sqrtf(red[0]), 1e-12f);
  for (int i = threadIdx.x; i < len; i += 256) p[i] *= inv;
}

// ------------------------------ MCB stage 2 --------------------------------
__global__ void k_sketch2(const float* __restrict__ x, const int* __restrict__ h,
                          const int* __restrict__ sg, float* __restrict__ S, int n) {
  int idx = blockIdx.x * 256 + threadIdx.x;  // over B*n
  int b = idx / n, j = idx % n;
  atomicAdd(&S[(size_t)b * D2n + h[j]], x[idx] * (float)(2 * sg[j] - 1));
}

// z[b,m] = sum_i As2[b,i]*Bfold[b,(5m-i) mod D2], Bfold[k]=sum_{f<5} Bs2[(k+f)%D2]
// As2 row staged into LDS by the Tensor Data Mover (1D tile: 5000 dwords);
// the B-fold is computed by the VALU while the TDM streams A in.
__global__ void k_conv2(const float* __restrict__ As2, const float* __restrict__ Bs2,
                        float* __restrict__ zb) {
  __shared__ float As[D2n];
  __shared__ float Bf[D2n];
  int b = blockIdx.x;
#if USE_TDM
  if (threadIdx.x < 32) {  // wave 0 issues the TDM copy As2[b,:] -> LDS As[]
    unsigned long long ga = (unsigned long long)(size_t)(const void*)(As2 + (size_t)b * D2n);
    unsigned lds = (unsigned)(size_t)(const void*)&As[0];  // low 32 bits of LDS ptr = byte offset
    // D# group0: count=1 | lds_addr | global_addr[56:0] | type=2
    u32x4 g0 = { 1u, lds, (unsigned)(ga & 0xFFFFFFFFu),
                 (unsigned)((ga >> 32) & 0x01FFFFFFu) | 0x80000000u };
    // D# group1: data_size=2 (4B); tensor_dim0 = tile_dim0 = stride = 5000; tensor_dim1=1
    i32x8 g1 = { 0x00020000,
                 (int)((D2n & 0xFFFF) << 16),
                 (int)((D2n >> 16) | (1 << 16)),
                 (int)((D2n & 0xFFFF) << 16),
                 0,
                 (int)D2n,
                 0, 0 };
    i32x4 g2 = { 0, 0, 0, 0 };
    i32x4 g3 = { 0, 0, 0, 0 };
#if defined(__clang_major__) && (__clang_major__ >= 23)
    i32x8 g4 = { 0, 0, 0, 0, 0, 0, 0, 0 };
    __builtin_amdgcn_tensor_load_to_lds(g0, g1, g2, g3, g4, 0);
#else
    __builtin_amdgcn_tensor_load_to_lds(g0, g1, g2, g3, 0);
#endif
    __builtin_amdgcn_s_wait_tensorcnt(0);
  }
  for (int k = threadIdx.x; k < D2n; k += 256) {
    float t = 0.f;
#pragma unroll
    for (int f = 0; f < 5; ++f) { int kk = k + f; if (kk >= D2n) kk -= D2n; t += Bs2[(size_t)b * D2n + kk]; }
    Bf[k] = t;
  }
#else
  for (int k = threadIdx.x; k < D2n; k += 256) {
    As[k] = As2[(size_t)b * D2n + k];
    float t = 0.f;
#pragma unroll
    for (int f = 0; f < 5; ++f) { int kk = k + f; if (kk >= D2n) kk -= D2n; t += Bs2[(size_t)b * D2n + kk]; }
    Bf[k] = t;
  }
#endif
  __syncthreads();
  for (int m = threadIdx.x; m < MOUTn; m += 256) {
    int base = m * 5;
    float sum = 0.f;
    for (int i = 0; i < D2n; ++i) {
      int a = base - i; if (a < 0) a += D2n;
      sum += As[i] * Bf[a];
    }
    zb[(size_t)b * MOUTn + m] = sum;
  }
}

// ------------------------------ final softmax ------------------------------
__global__ void k_softmax_out(const float* __restrict__ lg, float* __restrict__ out) {
  int b = blockIdx.x;
  const float* p = lg + (size_t)b * VOCn;
  float* q = out + (size_t)b * VOCn;
  __shared__ float red[256];
  float mx = -1e30f;
  for (int i = threadIdx.x; i < VOCn; i += 256) mx = fmaxf(mx, p[i]);
  red[threadIdx.x] = mx; __syncthreads();
  for (int o = 128; o > 0; o >>= 1) {
    if (threadIdx.x < o) red[threadIdx.x] = fmaxf(red[threadIdx.x], red[threadIdx.x + o]);
    __syncthreads();
  }
  mx = red[0]; __syncthreads();
  float s = 0.f;
  for (int i = threadIdx.x; i < VOCn; i += 256) s += __expf(p[i] - mx);
  red[threadIdx.x] = s; __syncthreads();
  for (int o = 128; o > 0; o >>= 1) {
    if (threadIdx.x < o) red[threadIdx.x] += red[threadIdx.x + o];
    __syncthreads();
  }
  float inv = 1.f / red[0];
  for (int i = threadIdx.x; i < VOCn; i += 256) q[i] = __expf(p[i] - mx) * inv;
}

// ---------------------------------------------------------------------------
extern "C" void kernel_launch(void* const* d_in, const int* in_sizes, int n_in,
                              void* d_out, int out_size, void* d_ws, size_t ws_size,
                              hipStream_t stream) {
  (void)in_sizes; (void)n_in; (void)out_size; (void)ws_size;
  const float* qe  = (const float*)d_in[0];
  const float* img = (const float*)d_in[1];
  const float* Wih = (const float*)d_in[2];
  const float* Whh = (const float*)d_in[3];
  const float* bih = (const float*)d_in[4];
  const float* bhh = (const float*)d_in[5];
  const float* Wq1 = (const float*)d_in[6];
  const float* bq1 = (const float*)d_in[7];
  const float* Wq2 = (const float*)d_in[8];
  const float* bq2 = (const float*)d_in[9];
  const float* Wi1 = (const float*)d_in[10];
  const float* bi1 = (const float*)d_in[11];
  const float* Wi2 = (const float*)d_in[12];
  const float* bi2 = (const float*)d_in[13];
  const float* Wp  = (const float*)d_in[14];
  const float* bp  = (const float*)d_in[15];
  const int* h1x = (const int*)d_in[16];
  const int* s1x = (const int*)d_in[17];
  const int* h1q = (const int*)d_in[18];
  const int* s1q = (const int*)d_in[19];
  const int* h2i = (const int*)d_in[20];
  const int* s2i = (const int*)d_in[21];
  const int* h2q = (const int*)d_in[22];
  const int* s2q = (const int*)d_in[23];
  float* out = (float*)d_out;

  // ---- f32 workspace pool (16B-aligned slices) ----
  float* p = (float*)d_ws;
  float* xW   = p; p += (size_t)Tn * Bn * G4n;       // 6,815,744
  float* hs   = p; p += (size_t)Tn * Bn * Hn;        // 1,703,936
  float* cst  = p; p += (size_t)Bn * Hn;
  float* gbuf = p; p += (size_t)Bn * G4n;
  float* qa1  = p; p += (size_t)Bn * Tn * 512;
  float* qatt = p; p += (size_t)Bn * 2 * Tn;
  float* qf   = p; p += (size_t)Bn * 2 * Hn;
  float* A1   = p; p += (size_t)Bn * D1n;            // 15,680,000
  float* Af   = p; p += (size_t)Bn * D1n;            // 15,680,000
  float* iq   = p; p += (size_t)Bn * MOUTn * Sn;
  float* ia1  = p; p += (size_t)Bn * Sn * 512;
  float* iatt = p; p += (size_t)Bn * 2 * Sn;
  float* ift  = p; p += (size_t)Bn * 2 * FEATn;
  float* As2  = p; p += (size_t)Bn * D2n;
  float* Bs2  = p; p += (size_t)Bn * D2n;
  float* zb   = p; p += (size_t)Bn * MOUTn;
  float* lg   = p; p += (size_t)Bn * VOCn;
  // ---- bf16 workspace pool ----
  unsigned short* u = (unsigned short*)p;
  unsigned short* qe_bf  = u; u += (size_t)(Tn * Bn) * EnP;   // 532,480
  unsigned short* Wih_bf = u; u += (size_t)G4n * EnP;         // 1,310,720
  unsigned short* Whh_bf = u; u += (size_t)G4n * Hn;          // 4,194,304
  unsigned short* hsA_bf = u; u += (size_t)Tn * Bn * Hn;      // 1,703,936
  unsigned short* hsB_bf = u; u += (size_t)Tn * Bn * Hn;      // 1,703,936
  unsigned short* hz_bf  = u; u += (size_t)Bn * Hn;           // 65,536
  unsigned short* Wq1_bf = u; u += (size_t)512 * Hn;
  unsigned short* iqT_bf = u; u += (size_t)(Bn * Sn) * MOUTP; // 3,211,264
  unsigned short* Wi1_bf = u; u += (size_t)512 * MOUTP;
  unsigned short* zb_bf  = u; u += (size_t)Bn * MOUTP;
  unsigned short* Wp_bf  = u; u += (size_t)VOCP * MOUTP;      // 3,080,192

  auto zero = [&](float* q0, size_t n) {
    k_zero<<<(unsigned)((n + 255) / 256), 256, 0, stream>>>(q0, n);
  };
  auto cvt = [&](const float* s, unsigned short* d, int rs, int K, int Kp, long tot) {
    k_cvt_bf<<<(unsigned)((tot + 255) / 256), 256, 0, stream>>>(s, d, rs, K, Kp, tot);
  };

  // ---- per-launch init + weight conversion ----
  zero(cst, (size_t)Bn * Hn);
  zero(A1,  (size_t)Bn * D1n);
  zero(As2, (size_t)Bn * D2n);
  zero(Bs2, (size_t)Bn * D2n);
  k_zero_us<<<(Bn * Hn) / 256, 256, 0, stream>>>(hz_bf, (size_t)Bn * Hn);
  k_cvt_qe<<<((Tn * Bn) * EnP) / 256, 256, 0, stream>>>(qe, qe_bf);
  cvt(Wih, Wih_bf, G4n, En, EnP, (long)G4n * EnP);
  cvt(Whh, Whh_bf, G4n, Hn, Hn, (long)G4n * Hn);
  cvt(Wq1, Wq1_bf, 512, Hn, Hn, (long)512 * Hn);
  cvt(Wi1, Wi1_bf, 512, MOUTn, MOUTP, (long)512 * MOUTP);
  cvt(Wp,  Wp_bf,  VOCn, MOUTn, MOUTP, (long)VOCP * MOUTP);

  // ---- LSTM ----
  // xW = qe @ Wih^T + bih + bhh   (1664 x 4096, K=320)
  k_gemm_bf4<<<dim3((Tn * Bn) / 16, G4n / 64), 32, 0, stream>>>(
      qe_bf, Wih_bf, xW, EnP, G4n, G4n, bih, bhh, 0);
  for (int t = 0; t < Tn; ++t) {
    const unsigned short* hp = (t == 0) ? hz_bf : (hsA_bf + (size_t)(t - 1) * Bn * Hn);
    k_gemm_bf4<<<dim3(Bn / 16, G4n / 64), 32, 0, stream>>>(
        hp, Whh_bf, gbuf, Hn, G4n, G4n, nullptr, nullptr, 0);
    k_gates<<<(Bn * Hn) / 256, 256, 0, stream>>>(xW, gbuf, cst, hs, hsA_bf, hsB_bf, t);
  }

  // ---- question attention ----
  k_gemm_bf4<<<dim3((Tn * Bn) / 16, 512 / 64), 32, 0, stream>>>(
      hsB_bf, Wq1_bf, qa1, Hn, 512, 512, bq1, nullptr, 1);
  k_att_logits<<<(Bn * 2 * Tn + 255) / 256, 256, 0, stream>>>(qa1, Wq2, bq2, qatt, Tn);
  k_softmax_small<<<1, 128, 0, stream>>>(qatt, Bn * 2, Tn);
  k_qfeat<<<(Bn * 2 * Hn) / 256, 256, 0, stream>>>(qatt, hs, qf);

  // ---- MCB1: sketch, fold, gather-conv, signed-sqrt + L2 norm ----
  k_sketch1<<<(unsigned)(((size_t)Bn * Sn * FEATn + 255) / 256), 256, 0, stream>>>(img, h1x, s1x, A1);
  k_fold1<<<(unsigned)(((size_t)Bn * D1n + 255) / 256), 256, 0, stream>>>(A1, Af);
  k_conv1<<<dim3(MOUTn, Bn), 64, 0, stream>>>(Af, qf, h1q, s1q, iq);
  k_ssqrt_l2<<<Bn, 256, 0, stream>>>(iq, MOUTn * Sn);

  // ---- image attention ----
  k_cvt_iqT<<<((Bn * Sn) * MOUTP) / 256, 256, 0, stream>>>(iq, iqT_bf);
  k_gemm_bf4<<<dim3((Bn * Sn) / 16, 512 / 64), 32, 0, stream>>>(
      iqT_bf, Wi1_bf, ia1, MOUTP, 512, 512, bi1, nullptr, 1);
  k_att_logits<<<(Bn * 2 * Sn + 255) / 256, 256, 0, stream>>>(ia1, Wi2, bi2, iatt, Sn);
  k_softmax_small<<<1, 128, 0, stream>>>(iatt, Bn * 2, Sn);
  k_ifeat<<<(Bn * 2 * FEATn) / 256, 256, 0, stream>>>(iatt, img, ift);

  // ---- MCB2: sketches, LDS circular conv (TDM-staged) with factor fold ----
  k_sketch2<<<(Bn * 2 * FEATn) / 256, 256, 0, stream>>>(ift, h2i, s2i, As2, 2 * FEATn);
  k_sketch2<<<(Bn * 2 * Hn) / 256, 256, 0, stream>>>(qf, h2q, s2q, Bs2, 2 * Hn);
  k_conv2<<<Bn, 256, 0, stream>>>(As2, Bs2, zb);
  k_ssqrt_l2<<<Bn, 256, 0, stream>>>(zb, MOUTn);

  // ---- classifier + softmax ----
  cvt(zb, zb_bf, Bn, MOUTn, MOUTP, (long)Bn * MOUTP);
  k_gemm_bf4<<<dim3(Bn / 16, VOCP / 64), 32, 0, stream>>>(
      zb_bf, Wp_bf, lg, MOUTP, VOCn, VOCn, bp, nullptr, 0);
  k_softmax_out<<<Bn, 256, 0, stream>>>(lg, out);
}
